// Transformer_68238440399084
// MI455X (gfx1250) — matrix operand order repaired and checked by reading, not compiled, not run on hardware
//
#include <hip/hip_runtime.h>

// ---------------- problem constants ----------------
constexpr int BB = 4;       // batch
constexpr int SE = 64;      // encoder seq len
constexpr int TM = 16;      // decoder max len
constexpr int DM = 512;     // model dim
constexpr int HH = 8;       // heads (each head full D->D)
constexpr int LL = 2;       // layers
constexpr int VV = 32000;   // zh vocab
constexpr int BOSTOK = 1;
constexpr float EPSV = 1e-5f;

typedef __attribute__((ext_vector_type(16))) _Float16 v16h;  // 8 VGPRs
typedef __attribute__((ext_vector_type(8)))  _Float16 v8h;   // 4 VGPRs
typedef __attribute__((ext_vector_type(4)))  _Float16 v4h;   // 2 VGPRs
typedef __attribute__((ext_vector_type(8)))  float    v8f;   // 8 VGPRs

// =====================================================================
// WMMA GEMM:  C[M,N] = act( A[M,K] @ W + bias + resid )
//   WMODE 0: W is flat [K,N] row-major
//   WMODE 1: W is per-head [N/512][K][512]; col n -> head n>>9
// Block = 128 threads = 4 waves. Block tile: 16 (M) x 128 (N).
// Each wave owns two 16x16 C tiles. K stepped by 32 with
// v_wmma_f32_16x16x32_f16; f32->f16 conversion on LDS staging.
// Software-pipelined: next tile's global loads are issued while the
// current tile's fragments feed the WMMAs.
// Requirements (guaranteed by call sites): K % 32 == 0, N % 128 == 0.
// M ragged handled via clamp + cndmask (loads) and guard (stores).
// LDS layouts give contiguous wide fragment loads:
//   As[m][k] row-major   -> a = 2x ds_load_b128 per lane
//   BsT[n][k] transposed -> b = 2x ds_load_b128 per lane per tile
// =====================================================================
#define GEMM_THREADS 128
#define BN 128

template <int WMODE>
__global__ __launch_bounds__(GEMM_THREADS)
void wmma_gemm_kernel(const float* __restrict__ A, const float* __restrict__ W,
                      const float* __restrict__ bias, const float* __restrict__ resid,
                      float* __restrict__ C, int M, int N, int K, int relu)
{
    __shared__ _Float16 As[16][32];    // 16x32 A tile (row-major, 64B rows)
    __shared__ _Float16 BsT[BN][32];   // B tile transposed: [n][k]

    const int tid  = threadIdx.x;
    const int wave = tid >> 5;
    const int lane = tid & 31;
    const int half = lane >> 4;   // 0: lanes 0-15, 1: lanes 16-31
    const int l16  = lane & 15;

    const int m0 = blockIdx.x * 16;
    const int n0 = blockIdx.y * BN;

    // ---- A staging geometry: 512 f32 per step = 1 float4 per thread ----
    const int a_m   = tid >> 3;        // 0..15
    const int a_kc  = (tid & 7) * 4;   // 0..28
    const int a_row = m0 + a_m;
    const int a_rc  = (a_row < M) ? a_row : (M - 1);
    const bool a_ok = (a_row < M);
    const float* aptr = A + (long)a_rc * K + a_kc;      // += 32 per step

    // ---- B staging geometry: 8 float4 per thread, fixed (k,n) slots ----
    const float* bptr[8];
    int b_nc[8];
    int b_k[8];
#pragma unroll
    for (int u = 0; u < 8; ++u) {
        int i  = tid + u * GEMM_THREADS;   // 0..1023
        int k  = i >> 5;                   // 0..31
        int nc = (i & 31) * 4;             // 0..124
        int gn = n0 + nc;
        b_nc[u] = nc;
        b_k[u]  = k;
        if (WMODE == 0) {
            bptr[u] = W + (long)k * N + gn;
        } else {
            int hd = gn >> 9;              // headDim = 512
            bptr[u] = W + ((long)hd * K + k) * 512 + (gn & 511);
        }
    }
    const long bstride = (WMODE == 0) ? (long)32 * N : (long)32 * 512;

    v8f acc0 = {};
    v8f acc1 = {};

    // ---- prologue: load first tile into registers ----
    float4 ra = *(const float4*)aptr;
    float4 rb[8];
#pragma unroll
    for (int u = 0; u < 8; ++u) rb[u] = *(const float4*)bptr[u];

    for (int k0 = 0; k0 < K; k0 += 32) {
        // ---- store staged registers into LDS (f32 -> f16) ----
        {
            v4h h;
            h[0] = (_Float16)(a_ok ? ra.x : 0.0f);
            h[1] = (_Float16)(a_ok ? ra.y : 0.0f);
            h[2] = (_Float16)(a_ok ? ra.z : 0.0f);
            h[3] = (_Float16)(a_ok ? ra.w : 0.0f);
            *(v4h*)&As[a_m][a_kc] = h;
        }
#pragma unroll
        for (int u = 0; u < 8; ++u) {
            BsT[b_nc[u] + 0][b_k[u]] = (_Float16)rb[u].x;
            BsT[b_nc[u] + 1][b_k[u]] = (_Float16)rb[u].y;
            BsT[b_nc[u] + 2][b_k[u]] = (_Float16)rb[u].z;
            BsT[b_nc[u] + 3][b_k[u]] = (_Float16)rb[u].w;
        }
        __syncthreads();

        // ---- issue next tile's global loads (overlap with WMMA) ----
        if (k0 + 32 < K) {
            aptr += 32;
            ra = *(const float4*)aptr;
#pragma unroll
            for (int u = 0; u < 8; ++u) {
                bptr[u] += bstride;
                rb[u] = *(const float4*)bptr[u];
            }
        }

        // ---- fragments: contiguous wide LDS loads ----
        // A per ISA 7.12.2: a[0..7]=As[m][half*8+0..7], a[8..15]=As[m][16+half*8+0..7]
        v8h alo = *(const v8h*)&As[l16][half * 8];
        v8h ahi = *(const v8h*)&As[l16][16 + half * 8];
        v16h a;
#pragma unroll
        for (int i = 0; i < 8; ++i) { a[i] = alo[i]; a[8 + i] = ahi[i]; }

        // B: b[0..7]=BsT[n][half*8+0..7], b[8..15]=BsT[n][16+half*8+0..7], N = lane%16
        v8h b0lo = *(const v8h*)&BsT[wave * 16 + l16][half * 8];
        v8h b0hi = *(const v8h*)&BsT[wave * 16 + l16][16 + half * 8];
        v8h b1lo = *(const v8h*)&BsT[64 + wave * 16 + l16][half * 8];
        v8h b1hi = *(const v8h*)&BsT[64 + wave * 16 + l16][16 + half * 8];
        v16h b0, b1;
#pragma unroll
        for (int i = 0; i < 8; ++i) { b0[i] = b0lo[i]; b0[8 + i] = b0hi[i]; b1[i] = b1lo[i]; b1[8 + i] = b1hi[i]; }

        acc0 = __builtin_amdgcn_wmma_f32_16x16x32_f16(false, a, false, b0,
                                                      (short)0, acc0, false, false);
        acc1 = __builtin_amdgcn_wmma_f32_16x16x32_f16(false, a, false, b1,
                                                      (short)0, acc1, false, false);
        __syncthreads();
    }

    // ---- epilogue (C/D layout: VGPR r -> M=r + half*8, N = lane%16) ----
#pragma unroll
    for (int t = 0; t < 2; ++t) {
        const v8f* accp = t ? &acc1 : &acc0;
        int gn = n0 + t * 64 + wave * 16 + l16;
#pragma unroll
        for (int r = 0; r < 8; ++r) {
            int gr = m0 + half * 8 + r;
            if (gr < M) {
                float v = (*accp)[r];
                if (bias)  v += bias[gn];
                if (resid) v += resid[(long)gr * N + gn];
                if (relu)  v = v > 0.0f ? v : 0.0f;
                C[(long)gr * N + gn] = v;
            }
        }
    }
}

// =====================================================================
// Elementwise / attention helper kernels (small FLOPs, plain VALU)
// =====================================================================
__device__ inline float pos_emb_val(int s, int d)
{
    float dv = __powf(10000.0f, -2.0f * (float)(d >> 1) / (float)DM);
    float x  = (float)s * dv;
    return (d & 1) ? __cosf(x) : __sinf(x);
}

__global__ void embed_enc_kernel(const int* __restrict__ pieces,
                                 const float* __restrict__ emb,
                                 float* __restrict__ out)
{
    int idx = blockIdx.x * blockDim.x + threadIdx.x;
    if (idx >= BB * SE * DM) return;
    int d = idx % DM, s = (idx / DM) % SE, b = idx / (DM * SE);
    int tok = pieces[b * SE + s];
    out[idx] = emb[(long)tok * DM + d] + pos_emb_val(s, d);
}

__global__ void embed_dec_kernel(const int* __restrict__ targets,
                                 const float* __restrict__ emb,
                                 float* __restrict__ out)
{
    int idx = blockIdx.x * blockDim.x + threadIdx.x;
    if (idx >= BB * TM * DM) return;
    int d = idx % DM, j = (idx / DM) % TM, b = idx / (DM * TM);
    int tok = (j == 0) ? BOSTOK : targets[b * TM + j];
    out[idx] = emb[(long)tok * DM + d] + pos_emb_val(j, d);
}

__global__ void dec_state_kernel(const float* __restrict__ femb,
                                 float* __restrict__ state, int t)
{
    int idx = blockIdx.x * blockDim.x + threadIdx.x;
    if (idx >= BB * TM * DM) return;
    int j = (idx / DM) % TM;
    state[idx] = (j < t) ? femb[idx] : 0.0f;
}

__global__ void extract_row_kernel(const float* __restrict__ state,
                                   float* __restrict__ rows, int t)
{
    int idx = blockIdx.x * blockDim.x + threadIdx.x;
    if (idx >= BB * DM) return;
    int d = idx % DM, b = idx / DM;
    rows[((long)b * (TM - 1) + (t - 1)) * DM + d] = state[((long)b * TM + t) * DM + d];
}

// q: [B,Tq,H,D]  k: [B,Tk,H,D]  s: [B,H,Tq,Tk]
__global__ void attn_scores_kernel(const float* __restrict__ q,
                                   const float* __restrict__ k,
                                   float* __restrict__ s,
                                   int Tq, int Tk, float scale, int maskT)
{
    int idx = blockIdx.x * blockDim.x + threadIdx.x;
    int total = BB * HH * Tq * Tk;
    if (idx >= total) return;
    int tk = idx % Tk;
    int tq = (idx / Tk) % Tq;
    int h  = (idx / (Tk * Tq)) % HH;
    int b  = idx / (Tk * Tq * HH);
    float out;
    if (maskT >= 0 && tk > maskT) {
        out = -1e9f;
    } else {
        const float4* qp = (const float4*)(q + ((long)(b * Tq + tq) * HH + h) * DM);
        const float4* kp = (const float4*)(k + ((long)(b * Tk + tk) * HH + h) * DM);
        float acc = 0.0f;
        for (int e = 0; e < DM / 4; ++e) {
            float4 a = qp[e], c = kp[e];
            acc += a.x * c.x + a.y * c.y + a.z * c.z + a.w * c.w;
        }
        out = acc * scale;
    }
    s[idx] = out;
}

__global__ void softmax_kernel(float* __restrict__ s, int rows, int len)
{
    int row = blockIdx.x * blockDim.x + threadIdx.x;
    if (row >= rows) return;
    float* p = s + (long)row * len;
    float m = p[0];
    for (int i = 1; i < len; ++i) m = fmaxf(m, p[i]);
    float sum = 0.0f;
    for (int i = 0; i < len; ++i) { float e = __expf(p[i] - m); p[i] = e; sum += e; }
    float inv = 1.0f / sum;
    for (int i = 0; i < len; ++i) p[i] *= inv;
}

// p: [B,H,Tq,Tk]  v: [B,Tk,H,D]  o: [B,Tq,H,D] (== concat layout [B,Tq,H*D])
__global__ void attn_av_kernel(const float* __restrict__ p,
                               const float* __restrict__ v,
                               float* __restrict__ o, int Tq, int Tk)
{
    int idx = blockIdx.x * blockDim.x + threadIdx.x;
    int total = BB * Tq * HH * DM;
    if (idx >= total) return;
    int e  = idx % DM;
    int h  = (idx / DM) % HH;
    int tq = (idx / (DM * HH)) % Tq;
    int b  = idx / (DM * HH * Tq);
    const float* pp = p + (((long)b * HH + h) * Tq + tq) * Tk;
    float acc = 0.0f;
    for (int tk = 0; tk < Tk; ++tk)
        acc += pp[tk] * v[((long)(b * Tk + tk) * HH + h) * DM + e];
    o[idx] = acc;
}

// out = LN(x + res) * g + b   (res optional), one 256-thread block per D=512 row
__global__ __launch_bounds__(256)
void layernorm_kernel(const float* __restrict__ x, const float* __restrict__ res,
                      const float* __restrict__ g, const float* __restrict__ b,
                      float* __restrict__ out, int rows)
{
    __shared__ float red[256];
    int row = blockIdx.x;
    int tid = threadIdx.x;
    long base = (long)row * DM;
    float v0 = x[base + tid]       + (res ? res[base + tid]       : 0.0f);
    float v1 = x[base + 256 + tid] + (res ? res[base + 256 + tid] : 0.0f);
    red[tid] = v0 + v1;
    __syncthreads();
    for (int o = 128; o > 0; o >>= 1) { if (tid < o) red[tid] += red[tid + o]; __syncthreads(); }
    float mu = red[0] / (float)DM;
    __syncthreads();
    float d0 = v0 - mu, d1 = v1 - mu;
    red[tid] = d0 * d0 + d1 * d1;
    __syncthreads();
    for (int o = 128; o > 0; o >>= 1) { if (tid < o) red[tid] += red[tid + o]; __syncthreads(); }
    float rstd = rsqrtf(red[0] / (float)DM + EPSV);
    out[base + tid]       = d0 * rstd * g[tid]       + b[tid];
    out[base + 256 + tid] = d1 * rstd * g[256 + tid] + b[256 + tid];
}

// =====================================================================
// Host orchestration
// =====================================================================
static inline dim3 ew_grid(int total) { return dim3((total + 255) / 256); }

extern "C" void kernel_launch(void* const* d_in, const int* in_sizes, int n_in,
                              void* d_out, int out_size, void* d_ws, size_t ws_size,
                              hipStream_t stream)
{
    (void)in_sizes; (void)n_in; (void)out_size; (void)ws_size;

    const int*   pieces  = (const int*)d_in[0];
    const int*   targets = (const int*)d_in[1];
    const float* en_emb  = (const float*)d_in[2];
    const float* zh_emb  = (const float*)d_in[3];

    const float *enc_Wq = (const float*)d_in[4],  *enc_bq = (const float*)d_in[5];
    const float *enc_Wk = (const float*)d_in[6],  *enc_bk = (const float*)d_in[7];
    const float *enc_Wv = (const float*)d_in[8],  *enc_bv = (const float*)d_in[9];
    const float *enc_Wo = (const float*)d_in[10], *enc_bo = (const float*)d_in[11];
    const float *dsWq = (const float*)d_in[12], *dsbq = (const float*)d_in[13];
    const float *dsWk = (const float*)d_in[14], *dsbk = (const float*)d_in[15];
    const float *dsWv = (const float*)d_in[16], *dsbv = (const float*)d_in[17];
    const float *dsWo = (const float*)d_in[18], *dsbo = (const float*)d_in[19];
    const float *dcWq = (const float*)d_in[20], *dcbq = (const float*)d_in[21];
    const float *dcWk = (const float*)d_in[22], *dcbk = (const float*)d_in[23];
    const float *dcWv = (const float*)d_in[24], *dcbv = (const float*)d_in[25];
    const float *dcWo = (const float*)d_in[26], *dcbo = (const float*)d_in[27];
    const float *eln1g = (const float*)d_in[28], *eln1b = (const float*)d_in[29];
    const float *eln2g = (const float*)d_in[30], *eln2b = (const float*)d_in[31];
    const float *dln1g = (const float*)d_in[32], *dln1b = (const float*)d_in[33];
    const float *dln2g = (const float*)d_in[34], *dln2b = (const float*)d_in[35];
    const float *dln3g = (const float*)d_in[36], *dln3b = (const float*)d_in[37];
    const float *eW1 = (const float*)d_in[38], *eb1 = (const float*)d_in[39];
    const float *eW2 = (const float*)d_in[40], *eb2 = (const float*)d_in[41];
    const float *dW1 = (const float*)d_in[42], *db1 = (const float*)d_in[43];
    const float *dW2 = (const float*)d_in[44], *db2 = (const float*)d_in[45];
    const float *out_W = (const float*)d_in[46], *out_b = (const float*)d_in[47];

    float* out = (float*)d_out;

    // ---- workspace carve (floats); total ~9.5M f32 (~38 MB) ----
    float* ws = (float*)d_ws;
    auto take = [&](long n) { float* p = ws; ws += n; return p; };
    const long BSD = (long)BB * SE * DM;     // 131072
    const long BTD = (long)BB * TM * DM;     // 32768
    float* full_emb   = take(BTD);
    float* enc_x      = take(BSD);
    float* enc_states = take((long)LL * BSD);
    float* qbuf       = take(BSD * HH);       // [B,S,H,D] max
    float* kbuf       = take(BSD * HH);
    float* vbuf       = take(BSD * HH);
    float* sbuf       = take((long)BB * HH * SE * SE);
    float* catbuf     = take(BSD * HH);
    float* abuf       = take(BSD);
    float* tmpa       = take(BSD);
    float* ffnbuf     = take(BSD);
    float* hbuf      = take(BSD);
    float* kcbuf      = take((long)LL * BSD * HH);  // cross K per layer
    float* vcbuf      = take((long)LL * BSD * HH);  // cross V per layer
    float* state      = take(BTD);
    float* astate     = take(BTD);
    float* cstate     = take(BTD);
    float* decrows    = take((long)(TM - 1) * BB * DM);

    auto gemm = [&](const float* A, const float* W, const float* bias,
                    const float* resid, float* C, int M, int N, int K,
                    int wmode, int relu) {
        dim3 grid((M + 15) / 16, N / BN);
        if (wmode == 0) {
            wmma_gemm_kernel<0><<<grid, GEMM_THREADS, 0, stream>>>(
                A, W, bias, resid, C, M, N, K, relu);
        } else {
            wmma_gemm_kernel<1><<<grid, GEMM_THREADS, 0, stream>>>(
                A, W, bias, resid, C, M, N, K, relu);
        }
    };

    const float scale = 1.0f / sqrtf((float)DM);
    const int ME = BB * SE;   // 256 encoder rows
    const int MDEC = BB * TM; // 64 decoder rows
    const long WQK = (long)HH * DM * DM;   // per-layer per-head weight block
    const long WO  = (long)HH * DM * DM;   // Wo per layer [H*D, D]

    // ================= encoder =================
    embed_enc_kernel<<<ew_grid(BB * SE * DM), 256, 0, stream>>>(pieces, en_emb, enc_x);
    const float* x = enc_x;
    for (int l = 0; l < LL; ++l) {
        gemm(x, enc_Wq + l * WQK, enc_bq + l * HH * DM, nullptr, qbuf, ME, HH * DM, DM, 1, 0);
        gemm(x, enc_Wk + l * WQK, enc_bk + l * HH * DM, nullptr, kbuf, ME, HH * DM, DM, 1, 0);
        gemm(x, enc_Wv + l * WQK, enc_bv + l * HH * DM, nullptr, vbuf, ME, HH * DM, DM, 1, 0);
        attn_scores_kernel<<<ew_grid(BB * HH * SE * SE), 256, 0, stream>>>(qbuf, kbuf, sbuf, SE, SE, scale, -1);
        softmax_kernel<<<ew_grid(BB * HH * SE), 256, 0, stream>>>(sbuf, BB * HH * SE, SE);
        attn_av_kernel<<<ew_grid(BB * SE * HH * DM), 256, 0, stream>>>(sbuf, vbuf, catbuf, SE, SE);
        gemm(catbuf, enc_Wo + l * WO, enc_bo + l * DM, nullptr, abuf, ME, DM, HH * DM, 0, 0);
        layernorm_kernel<<<ME, 256, 0, stream>>>(x, abuf, eln1g + l * DM, eln1b + l * DM, tmpa, ME);
        gemm(tmpa, eW1 + (long)l * DM * DM, eb1 + l * DM, nullptr, ffnbuf, ME, DM, DM, 0, 1);
        gemm(ffnbuf, eW2 + (long)l * DM * DM, eb2 + l * DM, nullptr, hbuf, ME, DM, DM, 0, 0);
        layernorm_kernel<<<ME, 256, 0, stream>>>(hbuf, nullptr, eln2g + l * DM, eln2b + l * DM,
                                                 enc_states + l * BSD, ME);
        x = enc_states + l * BSD;
    }

    // ================= decoder prep =================
    embed_dec_kernel<<<ew_grid(BB * TM * DM), 256, 0, stream>>>(targets, zh_emb, full_emb);
    // step-invariant cross-attention K/V projections of enc_states
    for (int l = 0; l < LL; ++l) {
        gemm(enc_states + l * BSD, dcWk + l * WQK, dcbk + l * HH * DM, nullptr,
             kcbuf + (long)l * BSD * HH, ME, HH * DM, DM, 1, 0);
        gemm(enc_states + l * BSD, dcWv + l * WQK, dcbv + l * HH * DM, nullptr,
             vcbuf + (long)l * BSD * HH, ME, HH * DM, DM, 1, 0);
    }

    // ================= teacher-forced decode steps =================
    for (int t = 1; t < TM; ++t) {
        dec_state_kernel<<<ew_grid(BB * TM * DM), 256, 0, stream>>>(full_emb, state, t);
        for (int l = 0; l < LL; ++l) {
            // masked self-attention over decoder prefix
            gemm(state, dsWq + l * WQK, dsbq + l * HH * DM, nullptr, qbuf, MDEC, HH * DM, DM, 1, 0);
            gemm(state, dsWk + l * WQK, dsbk + l * HH * DM, nullptr, kbuf, MDEC, HH * DM, DM, 1, 0);
            gemm(state, dsWv + l * WQK, dsbv + l * HH * DM, nullptr, vbuf, MDEC, HH * DM, DM, 1, 0);
            attn_scores_kernel<<<ew_grid(BB * HH * TM * TM), 256, 0, stream>>>(qbuf, kbuf, sbuf, TM, TM, scale, t);
            softmax_kernel<<<ew_grid(BB * HH * TM), 256, 0, stream>>>(sbuf, BB * HH * TM, TM);
            attn_av_kernel<<<ew_grid(BB * TM * HH * DM), 256, 0, stream>>>(sbuf, vbuf, catbuf, TM, TM);
            gemm(catbuf, dsWo + l * WO, dsbo + l * DM, nullptr, abuf, MDEC, DM, HH * DM, 0, 0);
            layernorm_kernel<<<MDEC, 256, 0, stream>>>(state, abuf, dln1g + l * DM, dln1b + l * DM, astate, MDEC);
            // cross-attention against encoder states
            gemm(astate, dcWq + l * WQK, dcbq + l * HH * DM, nullptr, qbuf, MDEC, HH * DM, DM, 1, 0);
            attn_scores_kernel<<<ew_grid(BB * HH * TM * SE), 256, 0, stream>>>(
                qbuf, kcbuf + (long)l * BSD * HH, sbuf, TM, SE, scale, -1);
            softmax_kernel<<<ew_grid(BB * HH * TM), 256, 0, stream>>>(sbuf, BB * HH * TM, SE);
            attn_av_kernel<<<ew_grid(BB * TM * HH * DM), 256, 0, stream>>>(
                sbuf, vcbuf + (long)l * BSD * HH, catbuf, TM, SE);
            gemm(catbuf, dcWo + l * WO, dcbo + l * DM, nullptr, abuf, MDEC, DM, HH * DM, 0, 0);
            layernorm_kernel<<<MDEC, 256, 0, stream>>>(astate, abuf, dln2g + l * DM, dln2b + l * DM, cstate, MDEC);
            // FFN (no residual around FFN, per source)
            gemm(cstate, dW1 + (long)l * DM * DM, db1 + l * DM, nullptr, ffnbuf, MDEC, DM, DM, 0, 1);
            gemm(ffnbuf, dW2 + (long)l * DM * DM, db2 + l * DM, nullptr, hbuf, MDEC, DM, DM, 0, 0);
            layernorm_kernel<<<MDEC, 256, 0, stream>>>(hbuf, nullptr, dln3g + l * DM, dln3b + l * DM, state, MDEC);
        }
        extract_row_kernel<<<ew_grid(BB * DM), 256, 0, stream>>>(state, decrows, t);
    }

    // ================= output head: one big GEMM over all steps =================
    // decrows rows ordered [b][t-1] -> C maps directly onto out[B, TM-1, VV]
    gemm(decrows, out_W, out_b, nullptr, out, BB * (TM - 1), VV, DM, 0, 0);
}
